// SNNQNet_29300266893854
// MI455X (gfx1250) — compile-verified, matrix operand order checked by reading, not run on hardware
//
#include <hip/hip_runtime.h>
#include <hip/hip_bf16.h>

// ---------------------------------------------------------------------------
// SNN forward (LIF encoder -> recurrent LIF -> LI readout -> softmax) on
// MI455X (gfx1250, wave32, WMMA 16x16x32 f16).
//
// Pipeline:
//   Kernel E: per-element 10-bit spike masks -> u16[B][2S] in workspace.
//   Kernel A: I_in[t] = spikes(t) @ w_in.T via v_wmma_f32_16x16x32_f16.
//             Two timesteps per block: mask words and B fragments are
//             reused across 8 WMMAs (4 N-tiles x 2 t), and all B fragments
//             are loaded before the WMMA burst so loads clause together.
//   Kernel B: 10-step recurrent scan; z@w_rec.T and z@w_out.T as WMMA with
//             weights in LDS; max-over-time + softmax in-block.
// ---------------------------------------------------------------------------

typedef __attribute__((ext_vector_type(16))) _Float16 v16h;
typedef __attribute__((ext_vector_type(8)))  float    v8f;

#define T_STEPS 10
#define B_TOT   4096
#define S_IN    1024
#define K_IN    2048   // 2*S
#define H_DIM   128
#define A_DIM   18

// ISA 7.12.2 16-bit A/B fragment: lane L holds k = j + (L>>4)*8 for the low
// 8 elements and k = 16 + j + (L>>4)*8 for the high 8. Contiguous 16B runs,
// so these merge into b128 loads.
__device__ __forceinline__ v16h load_frag16(const _Float16* base, int lane) {
  const _Float16* p = base + ((lane >> 4) << 3);
  v16h r;
#pragma unroll
  for (int j = 0; j < 8; ++j) { r[j] = p[j]; r[j + 8] = p[j + 16]; }
  return r;
}

__device__ __forceinline__ unsigned int spike_mask(float c) {
  float vv = 0.0f;
  unsigned int m = 0;
#pragma unroll
  for (int tt = 0; tt < T_STEPS; ++tt) {   // norse lif_current_encoder
    vv += 0.1f * (c - vv);                 // dt*tau_mem_inv = 0.1, v_leak = 0
    if (vv > 1.0f) { m |= (1u << tt); vv = 0.0f; }  // v_th=1, reset to 0
  }
  return m;
}

// ------------------- Kernel E: encode spike masks + cast w_in --------------
__global__ __launch_bounds__(256) void snn_encode(
    const float* __restrict__ x, unsigned short* __restrict__ masks) {
  int i = blockIdx.x * blockDim.x + threadIdx.x;   // i = b*1024 + s
  if (i >= B_TOT * S_IN) return;
  int b = i >> 10, s = i & (S_IN - 1);
  float xv = x[i];
  masks[(size_t)b * K_IN + s]        = (unsigned short)spike_mask(50.0f * fmaxf(xv, 0.0f));
  masks[(size_t)b * K_IN + S_IN + s] = (unsigned short)spike_mask(50.0f * fmaxf(-xv, 0.0f));
}

__global__ __launch_bounds__(256) void cast_w_in_f16(
    const float* __restrict__ w_in, _Float16* __restrict__ w_in_h) {
  for (int i = blockIdx.x * blockDim.x + threadIdx.x; i < H_DIM * K_IN;
       i += gridDim.x * blockDim.x)
    w_in_h[i] = (_Float16)w_in[i];
}

// --------------------------- Kernel A: input GEMM --------------------------
// grid = (B/64, T/2), block = 256 (8 waves). Block computes
// I_in[2y:2y+2, b:b+64, :]. Wave w: rows (w&3)*16, cols (w>>2)*64.
__global__ __launch_bounds__(256) void snn_input_gemm(
    const unsigned short* __restrict__ masks,
    const _Float16* __restrict__ w_in_h, float* __restrict__ i_in) {
  const int t0    = blockIdx.y * 2;
  const int bTile = blockIdx.x * 64;
  const int lane  = threadIdx.x & 31;
  const int wave  = threadIdx.x >> 5;
  const int mOff  = (wave & 3) * 16;
  const int nBase = (wave >> 2) * 64;
  const int row   = bTile + mOff + (lane & 15);   // A-frag row for this lane

  const unsigned short* mrow = masks + (size_t)row * K_IN;
  const _Float16 h1 = (_Float16)1.0f, h0 = (_Float16)0.0f;

  v8f acc[2][4] = {};
  for (int kc = 0; kc < K_IN; kc += 32) {
    // A fragments for steps t0, t0+1 from the same 16 mask words.
    const unsigned short* mp = mrow + kc + ((lane >> 4) << 3);
    v16h a0, a1;
#pragma unroll
    for (int j = 0; j < 8; ++j) {
      unsigned int mlo = mp[j], mhi = mp[j + 16];
      a0[j]     = ((mlo >> t0) & 1u)       ? h1 : h0;
      a0[j + 8] = ((mhi >> t0) & 1u)       ? h1 : h0;
      a1[j]     = ((mlo >> (t0 + 1)) & 1u) ? h1 : h0;
      a1[j + 8] = ((mhi >> (t0 + 1)) & 1u) ? h1 : h0;
    }
    // All 4 B fragments first (one clause, one wait), then the WMMA burst.
    v16h bf[4];
#pragma unroll
    for (int nt = 0; nt < 4; ++nt)
      bf[nt] = load_frag16(
          &w_in_h[(size_t)(nBase + nt * 16 + (lane & 15)) * K_IN + kc], lane);
#pragma unroll
    for (int nt = 0; nt < 4; ++nt) {
      acc[0][nt] = __builtin_amdgcn_wmma_f32_16x16x32_f16(
          false, a0, false, bf[nt], (short)0, acc[0][nt], false, false);
      acc[1][nt] = __builtin_amdgcn_wmma_f32_16x16x32_f16(
          false, a1, false, bf[nt], (short)0, acc[1][nt], false, false);
    }
  }
  // Store C tiles (lane L, vgpr r -> m = r + (L/16)*8, n = L&15).
#pragma unroll
  for (int dt = 0; dt < 2; ++dt)
#pragma unroll
    for (int nt = 0; nt < 4; ++nt)
#pragma unroll
      for (int r = 0; r < 8; ++r) {
        int m = mOff + r + ((lane >> 4) << 3);
        int n = nBase + nt * 16 + (lane & 15);
        i_in[((size_t)(t0 + dt) * B_TOT + bTile + m) * H_DIM + n] = acc[dt][nt][r];
      }
}

// --------------------------- Kernel B: recurrent scan ----------------------
// grid = B/32, block = 256 (8 waves). Block owns 32 batch rows x 128 neurons.
// Wave w: state tiles m=(w&1)*16, n=(w>>1)*32 (2 tiles). Waves 0-3 also do
// the LI readout (32 rows x 32 padded outputs, A=18 real).
__global__ __launch_bounds__(256) void snn_scan(
    const float* __restrict__ i_in, const float* __restrict__ w_rec,
    const float* __restrict__ w_out, float* __restrict__ out) {
  __shared__ _Float16 wrec_h[H_DIM * H_DIM];       // 32 KB, B[k][n]=w_rec[n][k]
  __shared__ _Float16 wout_h[32 * H_DIM];          //  8 KB, rows>=18 zero
  __shared__ _Float16 zbuf[2][32 * H_DIM];         // 16 KB ping-pong spikes
  __shared__ float    red[32 * 32];                //  4 KB reduction buffer

  const int tid   = threadIdx.x;
  const int lane  = tid & 31;
  const int wave  = tid >> 5;
  const int bTile = blockIdx.x * 32;

  for (int i = tid; i < H_DIM * H_DIM; i += 256) wrec_h[i] = (_Float16)w_rec[i];
  for (int i = tid; i < 32 * H_DIM; i += 256)
    wout_h[i] = ((i >> 7) < A_DIM) ? (_Float16)w_out[i] : (_Float16)0.0f;
  for (int i = tid; i < 32 * H_DIM; i += 256) zbuf[0][i] = (_Float16)0.0f;
  __syncthreads();

  const int mOff = (wave & 1) * 16;
  const int nB   = (wave >> 1) * 32;
  const int mHi  = (lane >> 4) << 3;
  const int nLoc = lane & 15;
  const bool isRO = (wave < 4);
  const int roM  = (wave & 1) * 16;
  const int roN  = (wave >> 1) * 16;   // only meaningful for waves 0-3

  v8f vstate[2] = {}, istate[2] = {};
  v8f vo = {}, io = {}, mx;
#pragma unroll
  for (int r = 0; r < 8; ++r) mx[r] = -3.0e38f;

  for (int t = 0; t < T_STEPS; ++t) {
    // --- elementwise LIF dynamics + spike emission into zbuf[(t+1)&1] ---
    v8f idec[2];
#pragma unroll
    for (int nt = 0; nt < 2; ++nt)
#pragma unroll
      for (int r = 0; r < 8; ++r) {
        int   m  = mOff + r + mHi;
        int   n  = nB + nt * 16 + nLoc;
        float I  = i_in[((size_t)t * B_TOT + bTile + m) * H_DIM + n];
        float vd = vstate[nt][r] + 0.1f * (istate[nt][r] - vstate[nt][r]);
        float id = 0.8f * istate[nt][r];               // 1 - dt*tau_syn_inv
        float zf = (vd > 1.0f) ? 1.0f : 0.0f;
        vstate[nt][r] = (1.0f - zf) * vd;
        idec[nt][r]   = id + I;
        zbuf[(t + 1) & 1][m * H_DIM + n] = (_Float16)zf;
      }
    __syncthreads();   // z_new complete; z_prev (zbuf[t&1]) free to read

    // --- i_new += z_prev @ w_rec.T (K=128, 4 WMMA k-steps per tile) ---
    v8f racc[2] = {};
#pragma unroll
    for (int kc = 0; kc < H_DIM; kc += 32) {
      v16h a = load_frag16(&zbuf[t & 1][(mOff + (lane & 15)) * H_DIM + kc], lane);
      v16h b0 = load_frag16(&wrec_h[(nB + (lane & 15)) * H_DIM + kc], lane);
      v16h b1 = load_frag16(&wrec_h[(nB + 16 + (lane & 15)) * H_DIM + kc], lane);
      racc[0] = __builtin_amdgcn_wmma_f32_16x16x32_f16(
          false, a, false, b0, (short)0, racc[0], false, false);
      racc[1] = __builtin_amdgcn_wmma_f32_16x16x32_f16(
          false, a, false, b1, (short)0, racc[1], false, false);
    }
#pragma unroll
    for (int nt = 0; nt < 2; ++nt)
#pragma unroll
      for (int r = 0; r < 8; ++r) istate[nt][r] = idec[nt][r] + racc[nt][r];

    // --- LI readout: i_jump = io + z_new @ w_out.T (waves 0-3) ---
    if (isRO) {
      v8f jacc = {};
#pragma unroll
      for (int kc = 0; kc < H_DIM; kc += 32) {
        v16h a = load_frag16(&zbuf[(t + 1) & 1][(roM + (lane & 15)) * H_DIM + kc], lane);
        v16h b = load_frag16(&wout_h[(roN + (lane & 15)) * H_DIM + kc], lane);
        jacc = __builtin_amdgcn_wmma_f32_16x16x32_f16(
            false, a, false, b, (short)0, jacc, false, false);
      }
#pragma unroll
      for (int r = 0; r < 8; ++r) {
        float ij  = io[r] + jacc[r];
        float von = vo[r] + 0.1f * (ij - vo[r]);
        io[r] = 0.8f * ij;
        vo[r] = von;
        mx[r] = fmaxf(mx[r], von);
      }
    }
    __syncthreads();   // protect zbuf[t&1] reads before next-step writes
  }

  // --- max-over-time is in mx; reduce to softmax over A=18 per row ---
  if (isRO) {
#pragma unroll
    for (int r = 0; r < 8; ++r)
      red[(roM + r + mHi) * 32 + roN + nLoc] = mx[r];
  }
  __syncthreads();
  if (tid < 32) {
    float m = -3.0e38f;
    for (int n = 0; n < A_DIM; ++n) m = fmaxf(m, red[tid * 32 + n]);
    float e[A_DIM], s = 0.0f;
    for (int n = 0; n < A_DIM; ++n) { e[n] = __expf(red[tid * 32 + n] - m); s += e[n]; }
    float inv = 1.0f / s;
    for (int n = 0; n < A_DIM; ++n)
      out[(size_t)(bTile + tid) * A_DIM + n] = e[n] * inv;
  }
}

// --------------------------- launcher --------------------------------------
extern "C" void kernel_launch(void* const* d_in, const int* in_sizes, int n_in,
                              void* d_out, int out_size, void* d_ws, size_t ws_size,
                              hipStream_t stream) {
  const float* x     = (const float*)d_in[0];   // [4096,1024]
  const float* w_in  = (const float*)d_in[1];   // [128,2048]
  const float* w_rec = (const float*)d_in[2];   // [128,128]
  const float* w_out = (const float*)d_in[3];   // [18,128]
  float* out = (float*)d_out;                   // [4096,18]

  char* ws = (char*)d_ws;
  size_t off = 0;
  float* i_in = (float*)(ws + off);             // 20 MB
  off += (size_t)T_STEPS * B_TOT * H_DIM * sizeof(float);
  _Float16* w_in_h = (_Float16*)(ws + off);     // 0.5 MB
  off += (size_t)H_DIM * K_IN * sizeof(_Float16);
  unsigned short* masks = (unsigned short*)(ws + off);   // 16 MB

  cast_w_in_f16<<<dim3(256), dim3(256), 0, stream>>>(w_in, w_in_h);
  snn_encode<<<dim3((B_TOT * S_IN) / 256), dim3(256), 0, stream>>>(x, masks);
  snn_input_gemm<<<dim3(B_TOT / 64, T_STEPS / 2), dim3(256), 0, stream>>>(masks, w_in_h, i_in);
  snn_scan<<<dim3(B_TOT / 32), dim3(256), 0, stream>>>(i_in, w_rec, w_out, out);
}
